// StackedMatchLSTM_3547642987249
// MI455X (gfx1250) — matrix operand order repaired and checked by reading, not compiled, not run on hardware
//
#include <hip/hip_runtime.h>

// ---------------------------------------------------------------------------
// StackedMatchLSTM for MI455X (gfx1250, wave32, WMMA).
// Strategy:
//  * One-time repack of all weight matrices into bf16 WMMA B-fragment layout
//    (32 contiguous bytes per lane per 32-deep k-tile) -> resident in 192MB L2
//    across the whole 128-step scan.
//  * All GEMMs have M=16 (batch) == WMMA tile M. One wave per 16-col N-tile,
//    v_wmma_f32_16x16x32_bf16 accumulation over K, fp32 A converted to bf16
//    fragments in-register.
//  * Attention (tanh/softmax/weighted-sum) and LayerNorm+LSTM cell fused into
//    block-per-batch VALU kernels with LDS reductions (wave32 shuffles).
// ---------------------------------------------------------------------------

typedef __bf16 bf16_t;
typedef __attribute__((ext_vector_type(16))) __bf16 v16bf;
typedef __attribute__((ext_vector_type(8)))  float  v8f;

#define EPSV 1e-6f

// ---------------------------------------------------------------------------
// Pack W (row-major K x N fp32, optionally two stacked sources W1 (K1 rows)
// then W2 (K2 rows)) into bf16 WMMA B-fragments.
// Tile order: [nt][kt], each tile = 32 lanes * 16 halves contiguous.
// Lane layout (contiguous k order, matching this file's A fragments): lanes 0-15 -> N=lane, K=kt*32+e (e=0..15);
// lanes 16-31 -> N=lane-16, K=kt*32+16+e.
// ---------------------------------------------------------------------------
__global__ __launch_bounds__(128)
void pack_b_frag(const float* __restrict__ W1, int K1,
                 const float* __restrict__ W2, int K2,
                 int N, bf16_t* __restrict__ out, int numTiles)
{
    const int lane = threadIdx.x & 31;
    const int wv   = threadIdx.x >> 5;
    const int KT   = (K1 + K2) >> 5;
    int tile = blockIdx.x * 4 + wv;
    if (tile >= numTiles) return;
    const int nt = tile / KT;
    const int kt = tile % KT;
    const int n  = nt * 16 + (lane & 15);
    const int kb = kt * 32 + ((lane < 16) ? 0 : 16);
    bf16_t* o = out + (size_t)tile * 512 + lane * 16;
#pragma unroll
    for (int e = 0; e < 16; ++e) {
        int k = kb + e;
        float v = (k < K1) ? W1[(size_t)k * N + n]
                           : W2[(size_t)(k - K1) * N + n];
        o[e] = (bf16_t)v;
    }
}

// ---------------------------------------------------------------------------
// C[16*mtiles x N] = concat(A1[16 x K1], A2[16 x K2]) @ Bpacked (+ bias).
// One wave per 16-wide N tile (grid.x * 4 waves), grid.y = m-tiles.
// A is fp32 row-major, converted to bf16 fragments in-register.
// A-fragment k order used here (contiguous, matches packed B): lanes 0-15 -> M=lane, elems 0..15 =
// K 0..15; lanes 16-31 -> M=lane-16, K 16..31.
// ---------------------------------------------------------------------------
__global__ __launch_bounds__(128)
void gemm16_bf16_wmma(const float* __restrict__ A1, int lda1, int K1,
                      const float* __restrict__ A2, int lda2, int K2,
                      const bf16_t* __restrict__ Bp,
                      const float* __restrict__ bias,
                      float* __restrict__ C, int ldc)
{
    const int lane = threadIdx.x & 31;
    const int wv   = threadIdx.x >> 5;
    const int nt   = blockIdx.x * 4 + wv;
    const int mt   = blockIdx.y;
    const int KT   = (K1 + K2) >> 5;
    const int m    = lane & 15;
    const int kbA  = (lane < 16) ? 0 : 16;

    const bf16_t* bptr = Bp + (size_t)nt * KT * 512 + lane * 16;
    v8f acc = {};

    for (int kt = 0; kt < KT; ++kt) {
        const int k0 = kt << 5;
        const float* arow;
        if (k0 < K1) arow = A1 + (size_t)(mt * 16 + m) * lda1 + k0 + kbA;
        else         arow = A2 + (size_t)(mt * 16 + m) * lda2 + (k0 - K1) + kbA;

        float4 x0 = *(const float4*)(arow);
        float4 x1 = *(const float4*)(arow + 4);
        float4 x2 = *(const float4*)(arow + 8);
        float4 x3 = *(const float4*)(arow + 12);

        v16bf a;
        a[0]  = (bf16_t)x0.x; a[1]  = (bf16_t)x0.y; a[2]  = (bf16_t)x0.z; a[3]  = (bf16_t)x0.w;
        a[4]  = (bf16_t)x1.x; a[5]  = (bf16_t)x1.y; a[6]  = (bf16_t)x1.z; a[7]  = (bf16_t)x1.w;
        a[8]  = (bf16_t)x2.x; a[9]  = (bf16_t)x2.y; a[10] = (bf16_t)x2.z; a[11] = (bf16_t)x2.w;
        a[12] = (bf16_t)x3.x; a[13] = (bf16_t)x3.y; a[14] = (bf16_t)x3.z; a[15] = (bf16_t)x3.w;

        v16bf b = *(const v16bf*)(bptr);
        bptr += 512;
        __builtin_prefetch(bptr, 0, 1);   // global_prefetch_b8: next B tile into L2/L0

        acc = __builtin_amdgcn_wmma_f32_16x16x32_bf16(
            /*neg_a=*/false, a, /*neg_b=*/false, b,
            /*c_mod=*/(short)0, acc, /*reuse_a=*/false, /*reuse_b=*/false);
    }

    const int n  = lane & 15;
    const int rb = (lane < 16) ? 0 : 8;
    const float bv = bias ? bias[nt * 16 + n] : 0.f;
    float* cp = C + (size_t)(mt * 16) * ldc + nt * 16 + n;
#pragma unroll
    for (int r = 0; r < 8; ++r)
        cp[(size_t)(rb + r) * ldc] = acc[r] + bv;
}

// ---------------------------------------------------------------------------
// Reductions (blockDim == 256, 8 waves of 32)
// ---------------------------------------------------------------------------
__device__ __forceinline__ float block_sum(float v, float* red)
{
    const int tid = threadIdx.x;
    const int lane = tid & 31, wv = tid >> 5;
#pragma unroll
    for (int o = 16; o > 0; o >>= 1) v += __shfl_xor(v, o, 32);
    if (lane == 0) red[wv] = v;
    __syncthreads();
    if (tid < 32) {
        float r = (lane < 8) ? red[lane] : 0.f;
#pragma unroll
        for (int o = 4; o > 0; o >>= 1) r += __shfl_xor(r, o, 32);
        if (lane == 0) red[0] = r;
    }
    __syncthreads();
    float out = red[0];
    __syncthreads();
    return out;
}

__device__ __forceinline__ float sigmoidf(float x) { return 1.f / (1.f + __expf(-x)); }

// ---------------------------------------------------------------------------
// Attention: per batch b (block): scores[sq] = w_att . tanh(qWq[b,sq,:]+s[b,:])
// masked softmax over 128, weighted = alpha @ input_q[b].
// ---------------------------------------------------------------------------
__global__ __launch_bounds__(256)
void attn_kernel(const float* __restrict__ qWq_l,   // [B,128,768]
                 const float* __restrict__ s,       // [B,768]
                 const float* __restrict__ w_att_l, // [768]
                 const float* __restrict__ mask_q,  // [B,128]
                 const float* __restrict__ input_q, // [B,128,768]
                 float* __restrict__ weighted)      // [B,768]
{
    __shared__ float s_sh[768];
    __shared__ float w_sh[768];
    __shared__ float sc[128];
    const int b = blockIdx.x;
    const int tid = threadIdx.x;
    const int lane = tid & 31, wv = tid >> 5;

    for (int j = tid; j < 768; j += 256) { s_sh[j] = s[b * 768 + j]; w_sh[j] = w_att_l[j]; }
    __syncthreads();

    // 8 waves x 16 rows = 128 score rows
#pragma unroll 1
    for (int r = 0; r < 16; ++r) {
        const int sq = wv * 16 + r;
        const float* g = qWq_l + ((size_t)b * 128 + sq) * 768;
        float p = 0.f;
        for (int a = lane; a < 768; a += 32) p += w_sh[a] * tanhf(g[a] + s_sh[a]);
#pragma unroll
        for (int o = 16; o > 0; o >>= 1) p += __shfl_xor(p, o, 32);
        if (lane == 0) {
            float mq = mask_q[b * 128 + sq];
            sc[sq] = (mq > 0.f) ? p : -1.0e9f;
        }
    }
    __syncthreads();

    if (tid < 32) {  // softmax over 128 by wave 0
        float mx = -1.0e30f;
        for (int i = lane; i < 128; i += 32) mx = fmaxf(mx, sc[i]);
#pragma unroll
        for (int o = 16; o > 0; o >>= 1) mx = fmaxf(mx, __shfl_xor(mx, o, 32));
        float sum = 0.f;
        for (int i = lane; i < 128; i += 32) { float e = __expf(sc[i] - mx); sc[i] = e; sum += e; }
#pragma unroll
        for (int o = 16; o > 0; o >>= 1) sum += __shfl_xor(sum, o, 32);
        float inv = 1.f / sum;
        for (int i = lane; i < 128; i += 32) sc[i] *= inv;
    }
    __syncthreads();

    for (int cidx = tid; cidx < 768; cidx += 256) {
        float acc = 0.f;
        const float* iq = input_q + (size_t)b * 128 * 768 + cidx;
#pragma unroll 4
        for (int sq = 0; sq < 128; ++sq) acc += sc[sq] * iq[(size_t)sq * 768];
        weighted[b * 768 + cidx] = acc;
    }
}

// ---------------------------------------------------------------------------
// LayerNorm(wi) + LayerNorm(wh) + bias -> LSTM cell -> LayerNorm(c) -> h1.
// One block per batch row. ddof=1 variance, matching the reference.
// ---------------------------------------------------------------------------
__global__ __launch_bounds__(256)
void gates_kernel(const float* __restrict__ wi_raw, const float* __restrict__ wh_raw,
                  const float* __restrict__ bias_l,
                  const float* __restrict__ gi, const float* __restrict__ bi,
                  const float* __restrict__ gh, const float* __restrict__ bh,
                  const float* __restrict__ gc, const float* __restrict__ bc,
                  const float* __restrict__ mask_p, int t,
                  float* __restrict__ h, float* __restrict__ c,
                  float* __restrict__ out_states, int write_out)
{
    __shared__ float pre_sh[3072];
    __shared__ float c_sh[768];
    __shared__ float red[8];
    const int b = blockIdx.x, tid = threadIdx.x;
    const float m = mask_p[(size_t)b * 128 + t];

    // LN(x @ W_ih)
    float xv[12];
    float sum = 0.f, sq = 0.f;
#pragma unroll
    for (int i = 0; i < 12; ++i) {
        float v = wi_raw[(size_t)b * 3072 + tid + i * 256];
        xv[i] = v; sum += v; sq += v * v;
    }
    float s1 = block_sum(sum, red);
    float s2 = block_sum(sq, red);
    float mean = s1 * (1.f / 3072.f);
    float var  = (s2 - s1 * mean) * (1.f / 3071.f);
    float inv  = 1.f / (sqrtf(var + EPSV) + EPSV);
#pragma unroll
    for (int i = 0; i < 12; ++i) {
        int j = tid + i * 256;
        pre_sh[j] = (gi[j] * (xv[i] - mean) * inv + bi[j]) * m;
    }

    // LN(h @ W_hh), accumulate + bias
    sum = 0.f; sq = 0.f;
#pragma unroll
    for (int i = 0; i < 12; ++i) {
        float v = wh_raw[(size_t)b * 3072 + tid + i * 256];
        xv[i] = v; sum += v; sq += v * v;
    }
    s1 = block_sum(sum, red);
    s2 = block_sum(sq, red);
    mean = s1 * (1.f / 3072.f);
    var  = (s2 - s1 * mean) * (1.f / 3071.f);
    inv  = 1.f / (sqrtf(var + EPSV) + EPSV);
#pragma unroll
    for (int i = 0; i < 12; ++i) {
        int j = tid + i * 256;
        pre_sh[j] += (gh[j] * (xv[i] - mean) * inv + bh[j]) * m + bias_l[j];
    }
    __syncthreads();

    // cell update (f,i,o,g split at 768 boundaries)
    float csum = 0.f, csq = 0.f;
#pragma unroll
    for (int i = 0; i < 3; ++i) {
        int u = tid + i * 256;
        float fg = pre_sh[u];
        float ig = pre_sh[768 + u];
        float gg = pre_sh[2304 + u];
        float cp = c[(size_t)b * 768 + u];
        float c1 = sigmoidf(fg) * cp + sigmoidf(ig) * tanhf(gg);
        c1 = c1 * m + cp * (1.f - m);
        c_sh[u] = c1;
        csum += c1; csq += c1 * c1;
    }
    s1 = block_sum(csum, red);
    s2 = block_sum(csq, red);
    float meanc = s1 * (1.f / 768.f);
    float varc  = (s2 - s1 * meanc) * (1.f / 767.f);
    float invc  = 1.f / (sqrtf(varc + EPSV) + EPSV);
#pragma unroll
    for (int i = 0; i < 3; ++i) {
        int u = tid + i * 256;
        float c1  = c_sh[u];
        float lnc = (gc[u] * (c1 - meanc) * invc + bc[u]) * m;
        float og  = pre_sh[1536 + u];
        float hp  = h[(size_t)b * 768 + u];
        float h1  = sigmoidf(og) * tanhf(lnc);
        h1 = h1 * m + hp * (1.f - m);
        h[(size_t)b * 768 + u] = h1;
        c[(size_t)b * 768 + u] = c1;
        if (write_out) out_states[((size_t)b * 128 + t) * 768 + u] = h1;
    }
}

__global__ __launch_bounds__(256)
void init_kernel(float* h, float* c, const float* mask_p, float* out_mask,
                 int nhc, int nmask)
{
    int i = blockIdx.x * 256 + threadIdx.x;
    if (i < nhc)   { h[i] = 0.f; c[i] = 0.f; }
    if (i < nmask) out_mask[i] = mask_p[i];
}

// ---------------------------------------------------------------------------
extern "C" void kernel_launch(void* const* d_in, const int* in_sizes, int n_in,
                              void* d_out, int out_size, void* d_ws, size_t ws_size,
                              hipStream_t stream)
{
    const float* input_p = (const float*)d_in[0];
    const float* mask_p  = (const float*)d_in[1];
    const float* input_q = (const float*)d_in[2];
    const float* mask_q  = (const float*)d_in[3];
    const float* W_ih    = (const float*)d_in[4];
    const float* W_hh    = (const float*)d_in[5];
    const float* bias    = (const float*)d_in[6];
    const float* ln_i_g  = (const float*)d_in[7];
    const float* ln_i_b  = (const float*)d_in[8];
    const float* ln_h_g  = (const float*)d_in[9];
    const float* ln_h_b  = (const float*)d_in[10];
    const float* ln_c_g  = (const float*)d_in[11];
    const float* ln_c_b  = (const float*)d_in[12];
    const float* Wq      = (const float*)d_in[13];
    const float* Wp      = (const float*)d_in[14];
    const float* Wr      = (const float*)d_in[15];
    const float* w_att   = (const float*)d_in[16];
    const float* b_att   = (const float*)d_in[17];
    float* out = (float*)d_out;

    const int L = 2, B = 16, T = 128, TQ = 128, P = 768, Q = 768, H = 768, A = 768;
    const int TIH = (1536 / 32) * (3072 / 16);  // 9216 tiles
    const int THH = (768  / 32) * (3072 / 16);  // 4608
    const int TPR = (1536 / 32) * (768  / 16);  // 2304 (stacked Wp;Wr)
    const int TWQ = (768  / 32) * (768  / 16);  // 1152

    char* ws = (char*)d_ws;
    size_t off = 0;
    auto alloc = [&](size_t bytes) -> char* {
        char* p = ws + off;
        off += (bytes + 255) & ~(size_t)255;
        return p;
    };

    bf16_t* wih_pk = (bf16_t*)alloc((size_t)L * TIH * 512 * 2);
    bf16_t* whh_pk = (bf16_t*)alloc((size_t)L * THH * 512 * 2);
    bf16_t* wpr_pk = (bf16_t*)alloc((size_t)L * TPR * 512 * 2);
    bf16_t* wq_pk  = (bf16_t*)alloc((size_t)L * TWQ * 512 * 2);
    float*  qWq    = (float*)alloc((size_t)L * B * TQ * A * 4);
    float*  hbuf   = (float*)alloc((size_t)L * B * H * 4);
    float*  cbuf   = (float*)alloc((size_t)L * B * H * 4);
    float*  sbuf   = (float*)alloc((size_t)B * A * 4);
    float*  wbuf   = (float*)alloc((size_t)B * Q * 4);
    float*  wi_raw = (float*)alloc((size_t)B * 4 * H * 4);
    float*  wh_raw = (float*)alloc((size_t)B * 4 * H * 4);

    // One-time weight repack into WMMA B-fragment layout (stays hot in L2).
    for (int l = 0; l < L; ++l) {
        pack_b_frag<<<TIH / 4, 128, 0, stream>>>(
            W_ih + (size_t)l * 1536 * 3072, 1536, nullptr, 0, 3072,
            wih_pk + (size_t)l * TIH * 512, TIH);
        pack_b_frag<<<THH / 4, 128, 0, stream>>>(
            W_hh + (size_t)l * 768 * 3072, 768, nullptr, 0, 3072,
            whh_pk + (size_t)l * THH * 512, THH);
        pack_b_frag<<<TPR / 4, 128, 0, stream>>>(
            Wp + (size_t)l * 768 * 768, 768,
            Wr + (size_t)l * 768 * 768, 768, 768,
            wpr_pk + (size_t)l * TPR * 512, TPR);
        pack_b_frag<<<TWQ / 4, 128, 0, stream>>>(
            Wq + (size_t)l * 768 * 768, 768, nullptr, 0, 768,
            wq_pk + (size_t)l * TWQ * 512, TWQ);
    }

    // h0=c0=0; emit mask part of the output tuple.
    init_kernel<<<(L * B * H + 255) / 256, 256, 0, stream>>>(
        hbuf, cbuf, mask_p, out + (size_t)B * T * H, L * B * H, B * T);

    // qWq[l] = input_q (2048x768) @ Wq[l]  -- grid.y = 128 m-tiles.
    for (int l = 0; l < L; ++l) {
        gemm16_bf16_wmma<<<dim3(12, 128), 128, 0, stream>>>(
            input_q, Q, 768, nullptr, 0, 0,
            wq_pk + (size_t)l * TWQ * 512, nullptr,
            qWq + (size_t)l * B * TQ * A, A);
    }

    // Sequential scan over time, layers inside.
    for (int t = 0; t < T; ++t) {
        for (int d = 0; d < L; ++d) {
            const float* cur;
            int lda_cur;
            if (d == 0) { cur = input_p + (size_t)t * P; lda_cur = T * P; }
            else        { cur = hbuf;                    lda_cur = H;     }
            float* h_d = hbuf + (size_t)d * B * H;
            float* c_d = cbuf + (size_t)d * B * H;

            // s = cur@Wp + h@Wr + b_att  (fused via stacked [Wp;Wr] packing)
            gemm16_bf16_wmma<<<dim3(12, 1), 128, 0, stream>>>(
                cur, lda_cur, 768, h_d, H, 768,
                wpr_pk + (size_t)d * TPR * 512,
                b_att + (size_t)d * A, sbuf, A);

            attn_kernel<<<16, 256, 0, stream>>>(
                qWq + (size_t)d * B * TQ * A, sbuf,
                w_att + (size_t)d * A, mask_q, input_q, wbuf);

            // wi_raw = [cur, weighted] @ W_ih  (concat via two A segments)
            gemm16_bf16_wmma<<<dim3(48, 1), 128, 0, stream>>>(
                cur, lda_cur, 768, wbuf, Q, 768,
                wih_pk + (size_t)d * TIH * 512, nullptr, wi_raw, 4 * H);

            // wh_raw = h @ W_hh
            gemm16_bf16_wmma<<<dim3(48, 1), 128, 0, stream>>>(
                h_d, H, 768, nullptr, 0, 0,
                whh_pk + (size_t)d * THH * 512, nullptr, wh_raw, 4 * H);

            gates_kernel<<<16, 256, 0, stream>>>(
                wi_raw, wh_raw, bias + (size_t)d * 4 * H,
                ln_i_g + (size_t)d * 4 * H, ln_i_b + (size_t)d * 4 * H,
                ln_h_g + (size_t)d * 4 * H, ln_h_b + (size_t)d * 4 * H,
                ln_c_g + (size_t)d * H,     ln_c_b + (size_t)d * H,
                mask_p, t, h_d, c_d, out, (d == L - 1) ? 1 : 0);
        }
    }
}
